// SparseVanillaGNN_19911468384542
// MI455X (gfx1250) — matrix operand order repaired and checked by reading, not compiled, not run on hardware
//
#include <hip/hip_runtime.h>
#include <hip/hip_bf16.h>
#include <math.h>

typedef __attribute__((ext_vector_type(16))) _Float16 v16h;
typedef __attribute__((ext_vector_type(8)))  float    v8f;

#define IN_DIM  256
#define HID_DIM 64
#define OUT_DIM 40

// Convert 8 contiguous f32 at p into v[o..o+7] as f16 (two float4 loads).
__device__ __forceinline__ void cvt8(const float* __restrict__ p, v16h& v, int o) {
    float4 q0 = *(const float4*)p;
    float4 q1 = *(const float4*)(p + 4);
    v[o + 0] = (_Float16)q0.x; v[o + 1] = (_Float16)q0.y;
    v[o + 2] = (_Float16)q0.z; v[o + 3] = (_Float16)q0.w;
    v[o + 4] = (_Float16)q1.x; v[o + 5] = (_Float16)q1.y;
    v[o + 6] = (_Float16)q1.z; v[o + 7] = (_Float16)q1.w;
}

// ---------------------------------------------------------------------------
// One-time f32 -> f16 weight conversion into workspace.
// ---------------------------------------------------------------------------
__global__ __launch_bounds__(256)
void k_cvt_w(const float* __restrict__ src, _Float16* __restrict__ dst, int n)
{
    int i = blockIdx.x * blockDim.x + threadIdx.x;
    if (i < n) dst[i] = (_Float16)src[i];
}

// ---------------------------------------------------------------------------
// GEMM1 + bias + ReLU:  h[N,64] = relu(x[N,256] @ W1[64,256]^T + b1)
// 8 waves / block; each wave: one 16-row tile x four 16-col tiles.
// A (x) converted f32->f16 on the fly; B (W1h) loaded directly as f16.
// ---------------------------------------------------------------------------
__global__ __launch_bounds__(256)
void k_gemm1(const float* __restrict__ x, const _Float16* __restrict__ W1h,
             const float* __restrict__ b1, float* __restrict__ h, int N)
{
    const int lane = threadIdx.x & 31;
    const int wave = threadIdx.x >> 5;
    const int l16  = lane & 15;
    const int hsel = lane >> 4;                 // 0: lanes 0-15, 1: lanes 16-31
    const int rowBase = blockIdx.x * 128 + wave * 16;

    int arow = rowBase + l16;
    if (arow >= N) arow = N - 1;                // clamp: loads safe, stores guarded
    const float* xrow = x + (size_t)arow * IN_DIM;

    v8f acc[4] = {v8f{}, v8f{}, v8f{}, v8f{}};

#pragma unroll
    for (int k0 = 0; k0 < IN_DIM; k0 += 32) {
        // A 16x32 f16: lane holds K = hsel*16 + [0..15]
        v16h a;
        cvt8(xrow + k0 + hsel * 16,     a, 0);
        cvt8(xrow + k0 + hsel * 16 + 8, a, 8);
#pragma unroll
        for (int t = 0; t < 4; ++t) {
            // B 32x16 f16: lane = column n; holds K = k0 + 16*hsel + [0..15]
            const v16h b = *(const v16h*)(W1h + (size_t)(t * 16 + l16) * IN_DIM
                                              + k0 + 16 * hsel);
            acc[t] = __builtin_amdgcn_wmma_f32_16x16x32_f16(
                false, a, false, b, (short)0, acc[t], false, false);
        }
    }

    // C/D layout: acc[t][r] = D[M = r + 8*hsel][n = l16] for col tile t
#pragma unroll
    for (int t = 0; t < 4; ++t) {
        const float bias = b1[t * 16 + l16];
#pragma unroll
        for (int r = 0; r < 8; ++r) {
            const int row = rowBase + r + 8 * hsel;
            if (row < N) {
                float v = acc[t][r] + bias;
                h[(size_t)row * HID_DIM + t * 16 + l16] = v > 0.f ? v : 0.f;
            }
        }
    }
}

// ---------------------------------------------------------------------------
// Zero agg buffer (float4 stores)
// ---------------------------------------------------------------------------
__global__ __launch_bounds__(256)
void k_zero(float4* __restrict__ p, int n4)
{
    int i = blockIdx.x * blockDim.x + threadIdx.x;
    if (i < n4) p[i] = make_float4(0.f, 0.f, 0.f, 0.f);
}

// ---------------------------------------------------------------------------
// Edge scatter: agg[dst] += h[src]; one wave per edge, float2 per lane.
// ---------------------------------------------------------------------------
__global__ __launch_bounds__(256)
void k_scatter(const int* __restrict__ ei, const float* __restrict__ h,
               float* __restrict__ agg, int E)
{
    const int gid  = blockIdx.x * blockDim.x + threadIdx.x;
    const int edge = gid >> 5;
    if (edge >= E) return;
    const int lane = gid & 31;
    const int dst = ei[edge];         // row (destination segment)
    const int src = ei[E + edge];     // col (neighbor to gather)
    const float2 v = *(const float2*)(h + (size_t)src * HID_DIM + lane * 2);
    float* a = agg + (size_t)dst * HID_DIM + lane * 2;
    atomicAdd(a,     v.x);
    atomicAdd(a + 1, v.y);
}

// ---------------------------------------------------------------------------
// GEMM2 + bias: out[N,40] = agg[N,64] @ W2[40,64]^T + b2   (cols 40..47 dropped)
// ---------------------------------------------------------------------------
__global__ __launch_bounds__(256)
void k_gemm2(const float* __restrict__ agg, const _Float16* __restrict__ W2h,
             const float* __restrict__ b2, float* __restrict__ out, int N)
{
    const int lane = threadIdx.x & 31;
    const int wave = threadIdx.x >> 5;
    const int l16  = lane & 15;
    const int hsel = lane >> 4;
    const int rowBase = blockIdx.x * 128 + wave * 16;

    int arow = rowBase + l16;
    if (arow >= N) arow = N - 1;
    const float* ar = agg + (size_t)arow * HID_DIM;

    v8f acc[3] = {v8f{}, v8f{}, v8f{}};

#pragma unroll
    for (int k0 = 0; k0 < HID_DIM; k0 += 32) {
        v16h a;
        cvt8(ar + k0 + hsel * 16,     a, 0);
        cvt8(ar + k0 + hsel * 16 + 8, a, 8);
#pragma unroll
        for (int t = 0; t < 3; ++t) {
            const int n = t * 16 + l16;
            v16h b = {};
            if (n < OUT_DIM)
                b = *(const v16h*)(W2h + (size_t)n * HID_DIM + k0 + 16 * hsel);
            acc[t] = __builtin_amdgcn_wmma_f32_16x16x32_f16(
                false, a, false, b, (short)0, acc[t], false, false);
        }
    }

#pragma unroll
    for (int t = 0; t < 3; ++t) {
        const int n = t * 16 + l16;
        if (n < OUT_DIM) {
            const float bias = b2[n];
#pragma unroll
            for (int r = 0; r < 8; ++r) {
                const int row = rowBase + r + 8 * hsel;
                if (row < N)
                    out[(size_t)row * OUT_DIM + n] = acc[t][r] + bias;
            }
        }
    }
}

// ---------------------------------------------------------------------------
// In-place row-wise log_softmax over 40 logits; one wave per row.
// ---------------------------------------------------------------------------
__global__ __launch_bounds__(256)
void k_logsoftmax(float* __restrict__ out, int N)
{
    const int row = blockIdx.x * 8 + (threadIdx.x >> 5);
    if (row >= N) return;
    const int lane = threadIdx.x & 31;
    float* p = out + (size_t)row * OUT_DIM;

    const bool hi = lane < (OUT_DIM - 32);   // lanes 0..7 also own col 32+lane
    float v0 = p[lane];
    float v1 = hi ? p[32 + lane] : -INFINITY;

    float m = fmaxf(v0, v1);
#pragma unroll
    for (int off = 16; off > 0; off >>= 1)
        m = fmaxf(m, __shfl_xor(m, off, 32));

    float s = expf(v0 - m) + (hi ? expf(v1 - m) : 0.f);
#pragma unroll
    for (int off = 16; off > 0; off >>= 1)
        s += __shfl_xor(s, off, 32);

    const float lse = m + logf(s);
    p[lane] = v0 - lse;
    if (hi) p[32 + lane] = v1 - lse;
}

// ---------------------------------------------------------------------------
extern "C" void kernel_launch(void* const* d_in, const int* in_sizes, int n_in,
                              void* d_out, int out_size, void* d_ws, size_t ws_size,
                              hipStream_t stream)
{
    const float* x  = (const float*)d_in[0];
    const int*   ei = (const int*)  d_in[1];
    const float* W1 = (const float*)d_in[2];
    const float* b1 = (const float*)d_in[3];
    const float* W2 = (const float*)d_in[4];
    const float* b2 = (const float*)d_in[5];
    float*       out = (float*)d_out;

    const int N = in_sizes[0] / IN_DIM;   // 100000
    const int E = in_sizes[1] / 2;        // 1600000

    // Workspace layout
    float*    h    = (float*)d_ws;                            // N*64 f32
    float*    agg  = h + (size_t)N * HID_DIM;                 // N*64 f32
    _Float16* W1h  = (_Float16*)(agg + (size_t)N * HID_DIM);  // 64*256 f16
    _Float16* W2h  = W1h + HID_DIM * IN_DIM;                  // 40*64  f16

    // One-time weight conversion (tiny)
    k_cvt_w<<<(HID_DIM * IN_DIM + 255) / 256, 256, 0, stream>>>(W1, W1h, HID_DIM * IN_DIM);
    k_cvt_w<<<(OUT_DIM * HID_DIM + 255) / 256, 256, 0, stream>>>(W2, W2h, OUT_DIM * HID_DIM);

    const int gemmBlocks = (N + 127) / 128;
    k_gemm1<<<gemmBlocks, 256, 0, stream>>>(x, W1h, b1, h, N);

    const int n4 = (N * HID_DIM) / 4;
    k_zero<<<(n4 + 255) / 256, 256, 0, stream>>>((float4*)agg, n4);

    const long long scThreads = (long long)E * 32;
    k_scatter<<<(int)((scThreads + 255) / 256), 256, 0, stream>>>(ei, h, agg, E);

    k_gemm2<<<gemmBlocks, 256, 0, stream>>>(agg, W2h, b2, out, N);

    k_logsoftmax<<<(N + 7) / 8, 256, 0, stream>>>(out, N);
}
